// GraphWaveNetLayer_7713761263797
// MI455X (gfx1250) — matrix operand order repaired and checked by reading, not compiled, not run on hardware
//
#include <hip/hip_runtime.h>
#include <math.h>

// ---------------------------------------------------------------------------
// GraphWaveNet layer for MI455X (gfx1250, wave32, WMMA + TDM).
//   x:[N,C,T] N=10000 C=64 T=12, convs kernel=2 causal, GCN over E=160000 edges
// Pipeline:
//   k_init    : agg=0, deg=1 (self loop)
//   k_deg     : deg[dst] += 1 (atomics)
//   k_norm    : norm[e] = rsqrt(deg[src])*rsqrt(deg[dst])
//   k_main    : TDM-staged x slab -> WMMA conv(F,G) -> gate -> WMMA @ gcn_w
//               -> hw[N*T,64]
//   k_scatter : agg[dst] += hw[src]*norm  (float4 gather, f32 atomic scatter)
//   k_final   : out[n,c,t] = agg + hw/deg + gcn_b + x
// ---------------------------------------------------------------------------

#define NN 10000
#define CC 64
#define TT 12
#define EE 160000
#define CT 768              // C*T per node
#define NT 120000           // N*T rows
#define NODES_PER_BLOCK 8   // 8 nodes * 12 t = 96 rows = 6 wave tiles
#define WAVES_PER_BLOCK 6
#define MAIN_THREADS (WAVES_PER_BLOCK * 32)
#define XS_ELEMS (NODES_PER_BLOCK * CT)   // 6144 f32, contiguous in memory

typedef __attribute__((ext_vector_type(16))) _Float16 v16h;
typedef __attribute__((ext_vector_type(8)))  float    v8f;
typedef __attribute__((ext_vector_type(4)))  unsigned int u32x4;
typedef __attribute__((ext_vector_type(8)))  int      i32x8;
typedef __attribute__((ext_vector_type(4)))  int      i32x4;

// ------------------------------ init ---------------------------------------
__global__ void k_init(float* __restrict__ agg, float* __restrict__ deg) {
    int i = blockIdx.x * blockDim.x + threadIdx.x;   // grid sized to NT*64
    agg[i] = 0.0f;
    if (i < NN) deg[i] = 1.0f;                       // self loop
}

// ------------------------------ degree -------------------------------------
__global__ void k_deg(const int* __restrict__ dst, float* __restrict__ deg) {
    int e = blockIdx.x * blockDim.x + threadIdx.x;   // grid sized to EE
    atomicAdd(&deg[dst[e]], 1.0f);
}

// ------------------------------ edge norm ----------------------------------
__global__ void k_norm(const int* __restrict__ src, const int* __restrict__ dst,
                       const float* __restrict__ deg, float* __restrict__ nrm) {
    int e = blockIdx.x * blockDim.x + threadIdx.x;   // grid sized to EE
    nrm[e] = rsqrtf(deg[src[e]]) * rsqrtf(deg[dst[e]]);
}

// --------------------- fused conv+gate+linear (WMMA) -----------------------
// Block owns NODES_PER_BLOCK consecutive nodes; their x slab is contiguous and
// staged into LDS by the Tensor Data Mover. Each wave owns one 16-row tile of
// rows (n*12+t); 6 waves cover the 96 rows exactly -> uniform control flow.
__launch_bounds__(MAIN_THREADS)
__global__ void k_main(const float* __restrict__ x,
                       const float* __restrict__ fw, const float* __restrict__ fb,
                       const float* __restrict__ gw, const float* __restrict__ gb,
                       const float* __restrict__ Wg,
                       float* __restrict__ hw_out) {
    // xs MUST be the first LDS object: TDM descriptor uses lds_addr = 0.
    __shared__ float    xs[XS_ELEMS];                       // 24 KB, offset 0
    __shared__ _Float16 sH[WAVES_PER_BLOCK][16][64];        // 12 KB

    const int lane  = threadIdx.x & 31;
    const int wave  = threadIdx.x >> 5;
    const int node0 = blockIdx.x * NODES_PER_BLOCK;
    const int lr    = lane & 15;     // A-row / B,C,D-column sub-index
    const int hi    = lane >> 4;     // lane half
    const int lrow  = wave * 16 + lr;          // local row in block (0..95)
    const int nloc  = lrow / TT;               // local node
    const int trow  = lrow - nloc * TT;        // timestep of this A-row
    const int grow0 = node0 * TT + wave * 16;  // first global row of tile

    // ---- Phase 0: TDM DMA of the contiguous x slab into LDS ---------------
    if (wave == 0) {
        unsigned long long ga =
            (unsigned long long)(const void*)(x + (size_t)node0 * CT);
        // D# group 0: count=1 | lds_addr=0 | global_addr | type=2 ("image")
        u32x4 g0 = { 1u, 0u, (unsigned)(ga & 0xffffffffu),
                     (unsigned)((ga >> 32) & 0x01ffffffu) | (2u << 30) };
        // D# group 1: data_size=4B; tensor_dim0=tile_dim0=6144; dim1=1;
        //             stride0=6144 (contiguous 1-row 2D copy)
        i32x8 g1 = { 0x00020000,                       // data_size=2 (4B)
                     (int)((unsigned)XS_ELEMS << 16),  // tensor_dim0 lo16<<16
                     0x00010000,                       // tensor_dim1=1
                     (int)((unsigned)XS_ELEMS << 16),  // tile_dim0<<16
                     1,                                // tile_dim1=1
                     XS_ELEMS,                         // tensor_dim0_stride
                     0, 0 };
        i32x4 z4 = { 0, 0, 0, 0 };                     // groups 2/3 unused (2D)
        i32x8 z8 = { 0, 0, 0, 0, 0, 0, 0, 0 };         // extra group (unused)
        __builtin_amdgcn_tensor_load_to_lds(g0, g1, z4, z4, z8, 0);
        __builtin_amdgcn_s_wait_tensorcnt(0);
    }
    __syncthreads();
    if (lane == 0) {                // warm GL2 with the (tiny, shared) weights
        __builtin_prefetch(fw, 0, 3);
        __builtin_prefetch(gw, 0, 3);
        __builtin_prefetch(Wg, 0, 3);
    }

    // ---- Phase 1: build conv A fragments (K=128 = tap*64+c) from LDS ------
    v16h aF[4];
#pragma unroll
    for (int kc = 0; kc < 4; ++kc) {
#pragma unroll
        for (int j = 0; j < 16; ++j) {
            // 16-bit A fragment K index (ISA 7.12.2 layout)
            int ka  = kc * 32 + ((j & 7) + ((j >> 3) << 4) + (hi << 3));
            int tap = ka >> 6, c = ka & 63;
            int tt  = trow + tap - 1;                  // causal: tap0 -> t-1
            aF[kc][j] = (tt >= 0) ? (_Float16)xs[nloc * CT + c * TT + tt]
                                  : (_Float16)0.0f;
        }
    }

    // ---- Phase 2: conv-as-matmul (M=16,K=128,N=64), gate, stage to LDS ----
    for (int nt = 0; nt < 4; ++nt) {
        const int o = nt * 16 + lr;                    // output channel column
        v8f accF = {}, accG = {};
#pragma unroll
        for (int kc = 0; kc < 4; ++kc) {
            v16h bF{}, bG{};
#pragma unroll
            for (int j = 0; j < 16; ++j) {
                int kb  = kc * 32 + ((j & 7) + ((j >> 3) << 4) + (hi << 3));     // B rows across lanes
                int tap = kb >> 6, c = kb & 63;
                bF[j] = (_Float16)fw[o * 128 + c * 2 + tap];
                bG[j] = (_Float16)gw[o * 128 + c * 2 + tap];
            }
            accF = __builtin_amdgcn_wmma_f32_16x16x32_f16(
                       false, aF[kc], false, bF, (short)0, accF, false, false);
            accG = __builtin_amdgcn_wmma_f32_16x16x32_f16(
                       false, aF[kc], false, bG, (short)0, accG, false, false);
        }
        const float biasF = fb[o], biasG = gb[o];
#pragma unroll
        for (int v = 0; v < 8; ++v) {
            // branch-free tanh / sigmoid via v_exp_f32 (saturate at +-inf)
            float z  = accF[v] + biasF;
            float th = 1.0f - 2.0f / (1.0f + __expf(2.0f * z));
            float sg = 1.0f / (1.0f + __expf(-(accG[v] + biasG)));
            int m = v + (hi << 3);                     // C/D row layout
            sH[wave][m][nt * 16 + lr] = (_Float16)(th * sg);
        }
    }
    __syncthreads();

    // ---- Phase 3: GCN linear  hw = H @ W   (M=16,K=64,N=64) ---------------
    v16h aH[2];
#pragma unroll
    for (int kc = 0; kc < 2; ++kc) {
#pragma unroll
        for (int j = 0; j < 16; ++j) {
            int ka = kc * 32 + ((j & 7) + ((j >> 3) << 4) + (hi << 3));
            aH[kc][j] = sH[wave][lr][ka];
        }
    }
    for (int nt = 0; nt < 4; ++nt) {
        v8f acc = {};
#pragma unroll
        for (int kc = 0; kc < 2; ++kc) {
            v16h b{};
#pragma unroll
            for (int j = 0; j < 16; ++j) {
                int kb = kc * 32 + ((j & 7) + ((j >> 3) << 4) + (hi << 3));
                b[j] = (_Float16)Wg[kb * 64 + nt * 16 + lr];
            }
            acc = __builtin_amdgcn_wmma_f32_16x16x32_f16(
                      false, aH[kc], false, b, (short)0, acc, false, false);
        }
#pragma unroll
        for (int v = 0; v < 8; ++v) {
            int m = v + (hi << 3);
            hw_out[(size_t)(grow0 + m) * 64 + nt * 16 + lr] = acc[v];
        }
    }
}

// ------------------------------ scatter-add --------------------------------
// One float4 of hw[src] per thread, 4 f32 atomics into agg[dst].
__global__ void k_scatter(const float* __restrict__ hw, const float* __restrict__ nrm,
                          const int* __restrict__ src, const int* __restrict__ dst,
                          float* __restrict__ agg) {
    long long i = (long long)blockIdx.x * blockDim.x + threadIdx.x;  // < E*192
    int e = (int)(i / 192);
    int q = (int)(i - (long long)e * 192) * 4;
    float nm = nrm[e];
    const float4 v = *(const float4*)(hw + (size_t)src[e] * CT + q);
    float* ap = agg + (size_t)dst[e] * CT + q;
    atomicAdd(ap + 0, v.x * nm);
    atomicAdd(ap + 1, v.y * nm);
    atomicAdd(ap + 2, v.z * nm);
    atomicAdd(ap + 3, v.w * nm);
}

// ------------------------------ finalize -----------------------------------
__global__ void k_final(const float* __restrict__ x, const float* __restrict__ agg,
                        const float* __restrict__ hw, const float* __restrict__ deg,
                        const float* __restrict__ gbias, float* __restrict__ out) {
    int i = blockIdx.x * blockDim.x + threadIdx.x;   // grid sized to N*C*T
    int n = i / CT;
    int rem = i - n * CT;
    int c = rem / TT;
    int t = rem - c * TT;
    size_t hidx = (size_t)n * CT + t * 64 + c;       // [N,T,C] element
    out[i] = agg[hidx] + hw[hidx] * (1.0f / deg[n]) + gbias[c] + x[i];
}

// ---------------------------------------------------------------------------
extern "C" void kernel_launch(void* const* d_in, const int* in_sizes, int n_in,
                              void* d_out, int out_size, void* d_ws, size_t ws_size,
                              hipStream_t stream) {
    const float* x  = (const float*)d_in[0];
    const float* fw = (const float*)d_in[1];
    const float* fb = (const float*)d_in[2];
    const float* gw = (const float*)d_in[3];
    const float* gb = (const float*)d_in[4];
    const float* Wg = (const float*)d_in[5];
    const float* gbias = (const float*)d_in[6];
    const int*   ei = (const int*)d_in[7];
    const int* src = ei;
    const int* dst = ei + EE;

    // workspace carve-up (floats): hw | agg | deg(pad 16384) | norm
    float* hw  = (float*)d_ws;
    float* agg = hw + (size_t)NT * 64;       // 7,680,000
    float* deg = agg + (size_t)NT * 64;      // 7,680,000
    float* nrm = deg + 16384;

    k_init   <<<NT * 64 / 256, 256, 0, stream>>>(agg, deg);
    k_deg    <<<EE / 256, 256, 0, stream>>>(dst, deg);
    k_norm   <<<EE / 256, 256, 0, stream>>>(src, dst, deg, nrm);
    k_main   <<<NN / NODES_PER_BLOCK, MAIN_THREADS, 0, stream>>>(
                 x, fw, fb, gw, gb, Wg, hw);
    k_scatter<<<(int)((long long)EE * 192 / 256), 256, 0, stream>>>(
                 hw, nrm, src, dst, agg);
    k_final  <<<NT * 64 / 256, 256, 0, stream>>>(x, agg, hw, deg, gbias,
                 (float*)d_out);
}
